// RNNPredictorBurnin_9878424781246
// MI455X (gfx1250) — matrix operand order repaired and checked by reading, not compiled, not run on hardware
//
#include <hip/hip_runtime.h>

// ---------------- problem constants ----------------
#define Hd   1024     // hidden
#define Od   1024     // output
#define Ad   2        // action dim
#define Bd   512      // batch
#define Td   64       // sequence length
#define G3   3072     // 3*H
#define KI   1056     // O+A padded up to a multiple of 32 (1026 -> 1056)

// ---------------- WMMA types ----------------
typedef __attribute__((ext_vector_type(16))) __bf16 v16bf;
typedef __attribute__((ext_vector_type(8)))  float  v8f;

__device__ __forceinline__ unsigned short f2bf(float f) {
    union { float f; unsigned u; } c; c.f = f;
    unsigned r = 0x7FFFu + ((c.u >> 16) & 1u);   // round-to-nearest-even
    return (unsigned short)((c.u + r) >> 16);
}

// ---------------- gfx1250 async global->LDS path ----------------
// Feature-detect async support via the builtin's existence, but emit the
// instruction with inline asm (portable across toolchains; the builtin's
// parameter LangAS ('__device__' int4*) is not spellable in HIP source).
#if __has_builtin(__builtin_amdgcn_global_load_async_to_lds_b128)
#define ASYNC_OK 1
#else
#define ASYNC_OK 0
#endif

#if ASYNC_OK
__device__ __forceinline__ void async_b128(const void* gsrc, void* ldst) {
    // vdst = LDS byte address (low 32 bits of generic LDS pointer),
    // vaddr = 64-bit global address, no SADDR.
    asm volatile("global_load_async_to_lds_b128 %0, %1, off"
                 :
                 : "v"((unsigned int)(uintptr_t)ldst),
                   "v"((unsigned long long)(uintptr_t)gsrc)
                 : "memory");
}
template <int N>
__device__ __forceinline__ void wait_async() {
#if __has_builtin(__builtin_amdgcn_s_wait_asynccnt)
    __builtin_amdgcn_s_wait_asynccnt(N);
#else
    asm volatile("s_wait_asynccnt %0" :: "n"(N) : "memory");
#endif
}
#endif

// ---------------- tiled bf16 WMMA GEMM ----------------
// C[M,N] = A[M,K] * W[N,K]^T + bias[N]
// A, W row-major bf16 (as u16). Optional bf16 mirror of C (Cbf).
// M,N divisible by 128; K divisible by 32.
#define BM 128
#define BN 128
#define BK 32
#define LSTR 40                 // LDS row stride (32 + 8 pad, keeps 16B alignment)
#define TILE_A (BM * LSTR)
#define TILE_W (BN * LSTR)
#if ASYNC_OK
#define NBUF 2
#else
#define NBUF 1
#endif

// 8 WMMAs on one staged K-tile: wave owns 32(M) x 64(N)
__device__ __forceinline__ void compute_tile(const unsigned short* Asb,
                                             const unsigned short* Wsb,
                                             v8f acc[2][4], int lane, int wm, int wn) {
    union Frag { uint4 u[2]; v16bf v; } fa[2], fb[4];
    // A 16x32 bf16 layout: lanes 0-15 K{0..7,16..23}; lanes 16-31 K{8..15,24..31}
    const int arow  = wm + (lane & 15);
    const int akoff = (lane < 16) ? 0 : 8;
    fa[0].u[0] = *reinterpret_cast<const uint4*>(&Asb[(arow)      * LSTR + akoff]);
    fa[0].u[1] = *reinterpret_cast<const uint4*>(&Asb[(arow)      * LSTR + akoff + 16]);
    fa[1].u[0] = *reinterpret_cast<const uint4*>(&Asb[(arow + 16) * LSTR + akoff]);
    fa[1].u[1] = *reinterpret_cast<const uint4*>(&Asb[(arow + 16) * LSTR + akoff + 16]);
    // B 32x16 bf16 layout: lane = N; lanes 0-15 K{0..7,16..23}; lanes 16-31 K{8..15,24..31}
    const int bcol  = wn + (lane & 15);
    const int bkoff = (lane < 16) ? 0 : 8;
#pragma unroll
    for (int tn = 0; tn < 4; ++tn) {
        fb[tn].u[0] = *reinterpret_cast<const uint4*>(&Wsb[(bcol + tn * 16) * LSTR + bkoff]);
        fb[tn].u[1] = *reinterpret_cast<const uint4*>(&Wsb[(bcol + tn * 16) * LSTR + bkoff + 16]);
    }
#pragma unroll
    for (int tm = 0; tm < 2; ++tm)
#pragma unroll
        for (int tn = 0; tn < 4; ++tn)
            acc[tm][tn] = __builtin_amdgcn_wmma_f32_16x16x32_bf16(
                false, fa[tm].v, false, fb[tn].v,
                (short)0, acc[tm][tn], false, false);
}

__device__ __forceinline__ void gemm_body(const unsigned short* __restrict__ A, int lda,
                                          const unsigned short* __restrict__ W, int ldw,
                                          float* __restrict__ C, int ldc,
                                          const float* __restrict__ bias,
                                          unsigned short* __restrict__ Cbf, int ldcbf,
                                          int K,
                                          unsigned short* As, unsigned short* Ws) {
    const int tid  = threadIdx.x;
    const int lane = tid & 31;
    const int wave = tid >> 5;
    const int wm   = (wave & 3) * 32;   // 4 waves along M
    const int wn   = (wave >> 2) * 64;  // 2 waves along N
    const int bm   = blockIdx.y * BM;
    const int bn   = blockIdx.x * BN;

    v8f acc[2][4] = {};

    // staging indices: 256 threads, each moves 2 A-chunks + 2 W-chunks (16B each)
    const int r0 = tid >> 2;            // 0..63
    const int c0 = (tid & 3) * 8;       // 0,8,16,24

#if ASYNC_OK
    auto issue_tile = [&](int k0, int buf) {
        unsigned short* Asb = As + buf * TILE_A;
        unsigned short* Wsb = Ws + buf * TILE_W;
        async_b128(&A[(size_t)(bm + r0)      * lda + k0 + c0], &Asb[(r0)      * LSTR + c0]);
        async_b128(&A[(size_t)(bm + r0 + 64) * lda + k0 + c0], &Asb[(r0 + 64) * LSTR + c0]);
        async_b128(&W[(size_t)(bn + r0)      * ldw + k0 + c0], &Wsb[(r0)      * LSTR + c0]);
        async_b128(&W[(size_t)(bn + r0 + 64) * ldw + k0 + c0], &Wsb[(r0 + 64) * LSTR + c0]);
    };
    const int ntiles = K / BK;
    issue_tile(0, 0);
    for (int i = 0; i < ntiles; ++i) {
        if (i + 1 < ntiles) {            // prefetch next tile into the other buffer
            issue_tile((i + 1) * BK, (i + 1) & 1);
            wait_async<4>();             // tile i landed; tile i+1 in flight
        } else {
            wait_async<0>();
        }
        __syncthreads();                 // whole workgroup sees tile i
        compute_tile(As + (i & 1) * TILE_A, Ws + (i & 1) * TILE_W, acc, lane, wm, wn);
        __syncthreads();                 // all waves done reading before buffer reuse
    }
#else
    for (int k0 = 0; k0 < K; k0 += BK) {
        const uint4 a0 = *reinterpret_cast<const uint4*>(&A[(size_t)(bm + r0)      * lda + k0 + c0]);
        const uint4 a1 = *reinterpret_cast<const uint4*>(&A[(size_t)(bm + r0 + 64) * lda + k0 + c0]);
        const uint4 w0 = *reinterpret_cast<const uint4*>(&W[(size_t)(bn + r0)      * ldw + k0 + c0]);
        const uint4 w1 = *reinterpret_cast<const uint4*>(&W[(size_t)(bn + r0 + 64) * ldw + k0 + c0]);
        __syncthreads();
        *reinterpret_cast<uint4*>(&As[(r0)      * LSTR + c0]) = a0;
        *reinterpret_cast<uint4*>(&As[(r0 + 64) * LSTR + c0]) = a1;
        *reinterpret_cast<uint4*>(&Ws[(r0)      * LSTR + c0]) = w0;
        *reinterpret_cast<uint4*>(&Ws[(r0 + 64) * LSTR + c0]) = w1;
        __syncthreads();
        compute_tile(As, Ws, acc, lane, wm, wn);
    }
#endif

    // epilogue: C/D 16x16 f32 layout: lane = N (mod 16); vgpr r -> M row; lanes 16-31 +8
    const int nin   = lane & 15;
    const int mhalf = (lane >> 4) * 8;
#pragma unroll
    for (int tm = 0; tm < 2; ++tm) {
#pragma unroll
        for (int tn = 0; tn < 4; ++tn) {
            const int col  = bn + wn + tn * 16 + nin;
            const float bv = bias ? bias[col] : 0.0f;
            const int rowb = bm + wm + tm * 16 + mhalf;
#pragma unroll
            for (int r = 0; r < 8; ++r) {
                float v = acc[tm][tn][r] + bv;
                C[(size_t)(rowb + r) * ldc + col] = v;
                if (Cbf) Cbf[(size_t)(rowb + r) * ldcbf + col] = f2bf(v);
            }
        }
    }
}

// single GEMM (used for the output projection, with bf16 feedback mirror)
__global__ __launch_bounds__(256)
void wmma_gemm_single(const unsigned short* __restrict__ A, int lda,
                      const unsigned short* __restrict__ W, int ldw,
                      float* __restrict__ C, int ldc,
                      const float* __restrict__ bias,
                      unsigned short* __restrict__ Cbf, int ldcbf, int K) {
    __shared__ unsigned short As[NBUF * TILE_A];
    __shared__ unsigned short Ws[NBUF * TILE_W];
    gemm_body(A, lda, W, ldw, C, ldc, bias, Cbf, ldcbf, K, As, Ws);
}

// fused pair of GEMMs (gi and gh), selected by blockIdx.z — halves launch count
__global__ __launch_bounds__(256)
void wmma_gemm_dual(const unsigned short* __restrict__ A0, int lda0,
                    const unsigned short* __restrict__ W0, int ldw0,
                    float* __restrict__ C0, const float* __restrict__ bias0, int K0,
                    const unsigned short* __restrict__ A1, int lda1,
                    const unsigned short* __restrict__ W1, int ldw1,
                    float* __restrict__ C1, const float* __restrict__ bias1, int K1,
                    int ldc) {
    __shared__ unsigned short As[NBUF * TILE_A];
    __shared__ unsigned short Ws[NBUF * TILE_W];
    if (blockIdx.z == 0)
        gemm_body(A0, lda0, W0, ldw0, C0, ldc, bias0, nullptr, 0, K0, As, Ws);
    else
        gemm_body(A1, lda1, W1, ldw1, C1, ldc, bias1, nullptr, 0, K1, As, Ws);
}

// ---------------- GRU gate fusion ----------------
__global__ __launch_bounds__(256)
void gru_gates_kernel(const float* __restrict__ gi, const float* __restrict__ gh,
                      float* __restrict__ h, unsigned short* __restrict__ hbf) {
    int idx = blockIdx.x * blockDim.x + threadIdx.x;
    if (idx >= Bd * Hd) return;
    int b = idx >> 10;
    int j = idx & 1023;
    const float* gib = gi + (size_t)b * G3;
    const float* ghb = gh + (size_t)b * G3;
    float r  = 1.0f / (1.0f + __expf(-(gib[j] + ghb[j])));
    float z  = 1.0f / (1.0f + __expf(-(gib[Hd + j] + ghb[Hd + j])));
    float n  = tanhf(gib[2 * Hd + j] + r * ghb[2 * Hd + j]);
    float hv = (1.0f - z) * n + z * h[idx];
    h[idx]   = hv;
    hbf[idx] = f2bf(hv);
}

// ---------------- per-step input packing ----------------
// Xi row: [ e(bf16, 1024 cols, written by out-GEMM) | a_t (2) | zero pad (30) ]
__global__ void pack_actions_kernel(const float* __restrict__ actions,
                                    unsigned short* __restrict__ Xi, int t) {
    int b = blockIdx.x * blockDim.x + threadIdx.x;
    if (b >= Bd) return;
    const float* a = actions + ((size_t)t * Bd + b) * Ad;
    unsigned short* row = Xi + (size_t)b * KI + Od;
    row[0] = f2bf(a[0]);
    row[1] = f2bf(a[1]);
#pragma unroll
    for (int i = Ad; i < KI - Od; ++i) row[i] = 0;
}

// ---------------- one-time conversions ----------------
__global__ void cvt_bf16_kernel(const float* __restrict__ src,
                                unsigned short* __restrict__ dst, int n) {
    int i = blockIdx.x * blockDim.x + threadIdx.x;
    if (i < n) dst[i] = f2bf(src[i]);
}

__global__ void pad_wih_kernel(const float* __restrict__ src,
                               unsigned short* __restrict__ dst) {
    int i = blockIdx.x * blockDim.x + threadIdx.x;
    if (i >= G3 * KI) return;
    int row = i / KI, col = i - row * KI;
    dst[i] = (col < (Od + Ad)) ? f2bf(src[(size_t)row * (Od + Ad) + col]) : (unsigned short)0;
}

__global__ void init_h_kernel(const float* __restrict__ h0, float* __restrict__ h,
                              unsigned short* __restrict__ hbf) {
    int i = blockIdx.x * blockDim.x + threadIdx.x;
    if (i >= Bd * Hd) return;
    float v = h0[i];
    h[i] = v;
    hbf[i] = f2bf(v);
}

__global__ void init_xi_kernel(const float* __restrict__ enc,
                               unsigned short* __restrict__ Xi) {
    int i = blockIdx.x * blockDim.x + threadIdx.x;
    if (i >= Bd * Od) return;
    int b = i >> 10, j = i & 1023;
    Xi[(size_t)b * KI + j] = f2bf(enc[i]);
}

// ---------------- launcher ----------------
extern "C" void kernel_launch(void* const* d_in, const int* in_sizes, int n_in,
                              void* d_out, int out_size, void* d_ws, size_t ws_size,
                              hipStream_t stream) {
    const float* enc     = (const float*)d_in[0];
    const float* actions = (const float*)d_in[1];
    const float* h0      = (const float*)d_in[2];
    const float* W_ih    = (const float*)d_in[3];
    const float* W_hh    = (const float*)d_in[4];
    const float* b_ih    = (const float*)d_in[5];
    const float* b_hh    = (const float*)d_in[6];
    const float* W_out   = (const float*)d_in[7];
    const float* b_out   = (const float*)d_in[8];
    float* out = (float*)d_out;

    char* ws = (char*)d_ws;
    size_t off = 0;
    auto suballoc = [&](size_t bytes) -> void* {
        void* p = ws + off;
        off = (off + bytes + 255) & ~(size_t)255;
        return p;
    };
    unsigned short* Xi   = (unsigned short*)suballoc((size_t)Bd * KI * 2);
    unsigned short* Wih  = (unsigned short*)suballoc((size_t)G3 * KI * 2);
    unsigned short* Whh  = (unsigned short*)suballoc((size_t)G3 * Hd * 2);
    unsigned short* Wout = (unsigned short*)suballoc((size_t)Od * Hd * 2);
    float*          gi   = (float*)suballoc((size_t)Bd * G3 * 4);
    float*          gh   = (float*)suballoc((size_t)Bd * G3 * 4);
    float*          h    = (float*)suballoc((size_t)Bd * Hd * 4);
    unsigned short* hbf  = (unsigned short*)suballoc((size_t)Bd * Hd * 2);

    // one-time conversions (bf16 weights stay resident in the 192MB L2)
    pad_wih_kernel<<<(G3 * KI + 255) / 256, 256, 0, stream>>>(W_ih, Wih);
    cvt_bf16_kernel<<<(G3 * Hd + 255) / 256, 256, 0, stream>>>(W_hh, Whh, G3 * Hd);
    cvt_bf16_kernel<<<(Od * Hd + 255) / 256, 256, 0, stream>>>(W_out, Wout, Od * Hd);
    init_h_kernel<<<(Bd * Hd + 255) / 256, 256, 0, stream>>>(h0, h, hbf);
    init_xi_kernel<<<(Bd * Od + 255) / 256, 256, 0, stream>>>(enc, Xi);

    const dim3 gridG(G3 / BN, Bd / BM, 2);  // fused gi+gh: 24 x 4 x 2
    const dim3 gridO(Od / BN, Bd / BM);     // out GEMM:     8 x 4

    for (int t = 0; t < Td; ++t) {
        pack_actions_kernel<<<2, 256, 0, stream>>>(actions, Xi, t);

        // z=0: gi = [e|a] @ W_ih^T + b_ih ; z=1: gh = h @ W_hh^T + b_hh
        wmma_gemm_dual<<<gridG, 256, 0, stream>>>(
            Xi,  KI, Wih, KI, gi, b_ih, KI,
            hbf, Hd, Whh, Hd, gh, b_hh, Hd,
            G3);

        gru_gates_kernel<<<(Bd * Hd + 255) / 256, 256, 0, stream>>>(gi, gh, h, hbf);

        // out_t = h @ W_out^T + b_out ; bf16 mirror feeds back into Xi as next e
        wmma_gemm_single<<<gridO, 256, 0, stream>>>(hbf, Hd, Wout, Hd,
                                                    out + (size_t)t * Bd * Od, Od,
                                                    b_out, Xi, KI, Hd);
    }
}
